// LocalEncoder_30734785970428
// MI455X (gfx1250) — matrix operand order repaired and checked by reading, not compiled, not run on hardware
//
#include <hip/hip_runtime.h>
#include <hip/hip_bf16.h>

// ---------------- problem constants ----------------
constexpr int B_    = 2;
constexpr int S_    = 4096;
constexpr int D_    = 1024;
constexpr int NH_   = 16;
constexpr int NKV_  = 4;
constexpr int HD_   = 64;
constexpr int W_    = 128;
constexpr int PS_   = 4;
constexpr int NP_   = S_ / PS_;       // 1024
constexpr int FFN_  = 2816;
constexpr int ROWS_ = B_ * S_;        // 8192
constexpr int CROWS_= B_ * NP_;       // 2048

typedef __bf16 bf16;
typedef __attribute__((ext_vector_type(16))) __bf16 v16bf;
typedef __attribute__((ext_vector_type(8)))  __bf16 v8bf;
typedef __attribute__((ext_vector_type(8)))  float  v8f;

// ---------------- fp32 -> bf16 transpose (weights: K x N -> N x K) -------
__global__ void k_conv_t(const float* __restrict__ W, bf16* __restrict__ Wt,
                         int K, int N) {
  int idx = blockIdx.x * 256 + threadIdx.x;
  if (idx >= K * N) return;
  int k = idx / N, n = idx % N;
  Wt[(size_t)n * K + k] = (bf16)W[idx];
}

// ---------------- embedding gather ----------------
__global__ void k_embed(const int* __restrict__ tokens,
                        const float* __restrict__ emb,
                        float* __restrict__ h) {
  int row = blockIdx.x;
  int tok = tokens[row];
  const float* src = emb + (size_t)tok * D_;
  float* dst = h + (size_t)row * D_;
  for (int d = threadIdx.x; d < D_; d += 256) dst[d] = src[d];
}

// ---------------- RMSNorm (fp32 in -> bf16 out), D = 1024 ----------------
__global__ __launch_bounds__(256) void k_rms(const float* __restrict__ x,
                                             const float* __restrict__ w,
                                             bf16* __restrict__ out) {
  int row = blockIdx.x;
  const float* xp = x + (size_t)row * D_;
  float vals[4];
  float loc = 0.f;
#pragma unroll
  for (int i = 0; i < 4; ++i) {
    float t = xp[threadIdx.x + i * 256];
    vals[i] = t;
    loc += t * t;
  }
#pragma unroll
  for (int off = 16; off > 0; off >>= 1) loc += __shfl_down(loc, off, 32);
  __shared__ float red[9];
  int wv = threadIdx.x >> 5, ln = threadIdx.x & 31;
  if (ln == 0) red[wv] = loc;
  __syncthreads();
  if (threadIdx.x == 0) {
    float t = 0.f;
#pragma unroll
    for (int i = 0; i < 8; ++i) t += red[i];
    red[8] = rsqrtf(t * (1.0f / D_) + 1e-5f);
  }
  __syncthreads();
  float rr = red[8];
  bf16* op = out + (size_t)row * D_;
#pragma unroll
  for (int i = 0; i < 4; ++i) {
    int d = threadIdx.x + i * 256;
    op[d] = (bf16)(vals[i] * rr * w[d]);
  }
}

// ---------------- RoPE in place on (rows, H, 64) fp32 ----------------
__global__ __launch_bounds__(128) void k_rope(float* __restrict__ x, int H, int HR) {
  int idx = blockIdx.x * 4 + (threadIdx.x >> 5);
  if (idx >= HR) return;
  int row = idx / H, hh = idx % H;
  int pos = row & (S_ - 1);
  int i = threadIdx.x & 31;
  // freq = 10000^(-i/32) = exp(-ln(1e4)/32 * i)
  float freq = __expf(-0.28782313662425572f * (float)i);
  float ang = (float)pos * freq;
  float s, c;
  __sincosf(ang, &s, &c);
  float* p = x + ((size_t)row * H + hh) * HD_ + 2 * i;
  float x1 = p[0], x2 = p[1];
  p[0] = x1 * c - x2 * s;
  p[1] = x1 * s + x2 * c;
}

// ---------------- patch max-pool: h (B,S,D) -> pooled bf16 (B,NP,D) ------
__global__ void k_pool(const float* __restrict__ h, bf16* __restrict__ pooled) {
  int pr = blockIdx.x;               // b*NP + p
  int b = pr >> 10, p = pr & (NP_ - 1);
  const float* base = h + ((size_t)b * S_ + (size_t)p * PS_) * D_;
  for (int d = threadIdx.x; d < D_; d += 256) {
    float m = base[d];
    m = fmaxf(m, base[D_ + d]);
    m = fmaxf(m, base[2 * D_ + d]);
    m = fmaxf(m, base[3 * D_ + d]);
    pooled[(size_t)pr * D_ + d] = (bf16)m;
  }
}

// ---------------- WMMA fragment builders ----------------
// A: row-major (M x K) bf16. 16-bit A 16x32 layout (matches frag_b):
//   lanes 0-15 : K 0..15 contiguous
//   lanes 16-31: K 16..31 contiguous
__device__ __forceinline__ v16bf frag_a(const bf16* __restrict__ A, int row,
                                        int K, int k0, int lane) {
  const bf16* p = A + (size_t)row * K + k0 + ((lane & 16) ? 16 : 0);
  return *(const v16bf*)p;
}
// B: stored column-major as Bt (N x K). 16-bit B 32x16, contiguous k order (same as frag_a, not the ISA table order):
//   lanes 0-15 : col = lane,    K = k0..k0+15 contiguous
//   lanes 16-31: col = lane-16, K = k0+16..k0+31
__device__ __forceinline__ v16bf frag_b(const bf16* __restrict__ Bt, int col,
                                        int K, int k0, int lane) {
  const bf16* p = Bt + (size_t)col * K + k0 + ((lane & 16) ? 16 : 0);
  return *(const v16bf*)p;
}

#define WMMA_BF16(A, Bf, C) \
  __builtin_amdgcn_wmma_f32_16x16x32_bf16(false, (A), false, (Bf), (short)0, (C), false, false)

// ---------------- generic WMMA GEMM: C(MxN) = A(MxK) * Bt(NxK)^T ---------
// EPI: 0 = store fp32, 1 = C += acc, 2 = store bf16, 3 = C = aux + acc
template <int EPI>
__global__ __launch_bounds__(256) void k_gemm(const bf16* __restrict__ A,
                                              const bf16* __restrict__ Bt,
                                              float* __restrict__ C,
                                              const float* __restrict__ aux,
                                              bf16* __restrict__ obf,
                                              int M, int N, int K) {
  const int lane = threadIdx.x & 31;
  const int wave = threadIdx.x >> 5;            // 8 waves
  const int rowBase = blockIdx.y * 128 + (wave >> 1) * 32;
  const int colBase = blockIdx.x * 64 + (wave & 1) * 32;
  v8f acc[2][2] = {};
  const int ra = rowBase + (lane & 15);
  const int ca = colBase + (lane & 15);
  for (int k0 = 0; k0 < K; k0 += 32) {
    v16bf a0 = frag_a(A, ra, K, k0, lane);
    v16bf a1 = frag_a(A, ra + 16, K, k0, lane);
    v16bf b0 = frag_b(Bt, ca, K, k0, lane);
    v16bf b1 = frag_b(Bt, ca + 16, K, k0, lane);
    acc[0][0] = WMMA_BF16(a0, b0, acc[0][0]);
    acc[0][1] = WMMA_BF16(a0, b1, acc[0][1]);
    acc[1][0] = WMMA_BF16(a1, b0, acc[1][0]);
    acc[1][1] = WMMA_BF16(a1, b1, acc[1][1]);
  }
  const int rowOff = (lane & 16) ? 8 : 0;
#pragma unroll
  for (int i = 0; i < 2; ++i)
#pragma unroll
    for (int j = 0; j < 2; ++j) {
      int col = colBase + j * 16 + (lane & 15);
#pragma unroll
      for (int e = 0; e < 8; ++e) {
        int row = rowBase + i * 16 + rowOff + e;
        size_t idx = (size_t)row * N + col;
        float val = acc[i][j][e];
        if (EPI == 0)      C[idx] = val;
        else if (EPI == 1) C[idx] += val;
        else if (EPI == 2) obf[idx] = (bf16)val;
        else               C[idx] = aux[idx] + val;
      }
    }
}

// ---------------- fused SwiGLU GEMM: out = silu(A*W1) * (A*W3) (bf16) ----
__global__ __launch_bounds__(256) void k_gemm_ffn(const bf16* __restrict__ A,
                                                  const bf16* __restrict__ B1t,
                                                  const bf16* __restrict__ B3t,
                                                  bf16* __restrict__ out,
                                                  int M, int N, int K) {
  const int lane = threadIdx.x & 31;
  const int wave = threadIdx.x >> 5;
  const int rowBase = blockIdx.y * 128 + (wave >> 1) * 32;
  const int colBase = blockIdx.x * 64 + (wave & 1) * 32;
  v8f ac1[2][2] = {};
  v8f ac3[2][2] = {};
  const int ra = rowBase + (lane & 15);
  const int ca = colBase + (lane & 15);
  for (int k0 = 0; k0 < K; k0 += 32) {
    v16bf a0 = frag_a(A, ra, K, k0, lane);
    v16bf a1 = frag_a(A, ra + 16, K, k0, lane);
    v16bf p0 = frag_b(B1t, ca, K, k0, lane);
    v16bf p1 = frag_b(B1t, ca + 16, K, k0, lane);
    v16bf q0 = frag_b(B3t, ca, K, k0, lane);
    v16bf q1 = frag_b(B3t, ca + 16, K, k0, lane);
    ac1[0][0] = WMMA_BF16(a0, p0, ac1[0][0]);
    ac1[0][1] = WMMA_BF16(a0, p1, ac1[0][1]);
    ac1[1][0] = WMMA_BF16(a1, p0, ac1[1][0]);
    ac1[1][1] = WMMA_BF16(a1, p1, ac1[1][1]);
    ac3[0][0] = WMMA_BF16(a0, q0, ac3[0][0]);
    ac3[0][1] = WMMA_BF16(a0, q1, ac3[0][1]);
    ac3[1][0] = WMMA_BF16(a1, q0, ac3[1][0]);
    ac3[1][1] = WMMA_BF16(a1, q1, ac3[1][1]);
  }
  const int rowOff = (lane & 16) ? 8 : 0;
#pragma unroll
  for (int i = 0; i < 2; ++i)
#pragma unroll
    for (int j = 0; j < 2; ++j) {
      int col = colBase + j * 16 + (lane & 15);
#pragma unroll
      for (int e = 0; e < 8; ++e) {
        int row = rowBase + i * 16 + rowOff + e;
        float x1 = ac1[i][j][e];
        float x3 = ac3[i][j][e];
        float sl = x1 / (1.0f + __expf(-x1));
        out[(size_t)row * N + col] = (bf16)(sl * x3);
      }
    }
}

// ---------------- sliding-window attention ----------------
// grid (NH, nb=32, B), block 128: one query row per thread, flash-style.
// q: (B,S,NH,64) fp32, k/v: (B,S,NKV,64) fp32, out bf16 (B,S,NH*64)
__global__ __launch_bounds__(128) void k_swa(const float* __restrict__ q,
                                             const float* __restrict__ k,
                                             const float* __restrict__ v,
                                             bf16* __restrict__ out) {
  const int h  = blockIdx.x;
  const int qb = blockIdx.y;
  const int b  = blockIdx.z;
  const int kvh = h >> 2;
  const int r = threadIdx.x;
  const int qpos = qb * W_ + r;
  float qreg[64];
  const float* qp = q + (((size_t)(b * S_ + qpos)) * NH_ + h) * HD_;
#pragma unroll
  for (int d = 0; d < 64; d += 4) {
    float4 t = *(const float4*)(qp + d);
    qreg[d] = t.x; qreg[d + 1] = t.y; qreg[d + 2] = t.z; qreg[d + 3] = t.w;
  }
  float o[64];
#pragma unroll
  for (int d = 0; d < 64; ++d) o[d] = 0.f;
  float m = -1e30f, l = 0.f;

  __shared__ float Ks[64][72];
  __shared__ float Vs[64][72];
  const int winStart = qb * W_ - W_;
  for (int c = 0; c < 6; ++c) {                // 6 * 64 = 384 keys
    int kbase = winStart + c * 64;
    int rrow = r >> 1;
    int dof = (r & 1) * 32;
    int kpos = kbase + rrow;
    bool in = (kpos >= 0 && kpos < S_);
    const float* kp = k + (((size_t)(b * S_ + (in ? kpos : 0))) * NKV_ + kvh) * HD_ + dof;
    const float* vp = v + (((size_t)(b * S_ + (in ? kpos : 0))) * NKV_ + kvh) * HD_ + dof;
    __syncthreads();
#pragma unroll
    for (int d = 0; d < 32; d += 4) {
      float4 tk = in ? *(const float4*)(kp + d) : make_float4(0.f, 0.f, 0.f, 0.f);
      float4 tv = in ? *(const float4*)(vp + d) : make_float4(0.f, 0.f, 0.f, 0.f);
      *(float4*)&Ks[rrow][dof + d] = tk;
      *(float4*)&Vs[rrow][dof + d] = tv;
    }
    __syncthreads();
    for (int key = 0; key < 64; ++key) {
      int kk = kbase + key;
      if (kk < 0 || kk >= S_) continue;
      int dd = kk - qpos;
      if (dd < -W_ || dd > W_) continue;
      float s = 0.f;
#pragma unroll
      for (int d = 0; d < 64; ++d) s += qreg[d] * Ks[key][d];
      s *= 0.125f;
      float nm = fmaxf(m, s);
      float corr = __expf(m - nm);
      float p = __expf(s - nm);
      l = l * corr + p;
#pragma unroll
      for (int d = 0; d < 64; ++d) o[d] = o[d] * corr + p * Vs[key][d];
      m = nm;
    }
  }
  float inv = 1.0f / l;
  bf16* op = out + (size_t)(b * S_ + qpos) * (NH_ * HD_) + h * HD_;
#pragma unroll
  for (int d = 0; d < 64; ++d) op[d] = (bf16)(o[d] * inv);
}

// ---------------- cross attention: 1024 queries vs 4096 keys -------------
// grid (NH, NP/128, B), block 128
__global__ __launch_bounds__(128) void k_xattn(const float* __restrict__ cq,
                                               const float* __restrict__ ck,
                                               const float* __restrict__ cv,
                                               bf16* __restrict__ out) {
  const int h  = blockIdx.x;
  const int qt = blockIdx.y;
  const int b  = blockIdx.z;
  const int kvh = h >> 2;
  const int r = threadIdx.x;
  const int qi = qt * 128 + r;
  float qreg[64];
  const float* qp = cq + (((size_t)(b * NP_ + qi)) * NH_ + h) * HD_;
#pragma unroll
  for (int d = 0; d < 64; d += 4) {
    float4 t = *(const float4*)(qp + d);
    qreg[d] = t.x; qreg[d + 1] = t.y; qreg[d + 2] = t.z; qreg[d + 3] = t.w;
  }
  float o[64];
#pragma unroll
  for (int d = 0; d < 64; ++d) o[d] = 0.f;
  float m = -1e30f, l = 0.f;

  __shared__ float Ks[64][72];
  __shared__ float Vs[64][72];
  for (int c = 0; c < S_ / 64; ++c) {
    int kbase = c * 64;
    int rrow = r >> 1;
    int dof = (r & 1) * 32;
    int kpos = kbase + rrow;
    const float* kp = ck + (((size_t)(b * S_ + kpos)) * NKV_ + kvh) * HD_ + dof;
    const float* vp = cv + (((size_t)(b * S_ + kpos)) * NKV_ + kvh) * HD_ + dof;
    __syncthreads();
#pragma unroll
    for (int d = 0; d < 32; d += 4) {
      *(float4*)&Ks[rrow][dof + d] = *(const float4*)(kp + d);
      *(float4*)&Vs[rrow][dof + d] = *(const float4*)(vp + d);
    }
    __syncthreads();
    for (int key = 0; key < 64; ++key) {
      float s = 0.f;
#pragma unroll
      for (int d = 0; d < 64; ++d) s += qreg[d] * Ks[key][d];
      s *= 0.125f;
      float nm = fmaxf(m, s);
      float corr = __expf(m - nm);
      float p = __expf(s - nm);
      l = l * corr + p;
#pragma unroll
      for (int d = 0; d < 64; ++d) o[d] = o[d] * corr + p * Vs[key][d];
      m = nm;
    }
  }
  float inv = 1.0f / l;
  bf16* op = out + (size_t)(b * NP_ + qi) * (NH_ * HD_) + h * HD_;
#pragma unroll
  for (int d = 0; d < 64; ++d) op[d] = (bf16)(o[d] * inv);
}

// =======================================================================
extern "C" void kernel_launch(void* const* d_in, const int* in_sizes, int n_in,
                              void* d_out, int out_size, void* d_ws, size_t ws_size,
                              hipStream_t stream) {
  // input order from setup_inputs()
  const int*   tokens      = (const int*)d_in[0];
  // d_in[1] = patch_ids (implicit arange(S)//PS, unused)
  const float* tok_emb     = (const float*)d_in[2];
  const float* attn_norm_w = (const float*)d_in[3];
  const float* wq          = (const float*)d_in[4];
  const float* wk          = (const float*)d_in[5];
  const float* wv          = (const float*)d_in[6];
  const float* wo          = (const float*)d_in[7];
  const float* ffn_norm_w  = (const float*)d_in[8];
  const float* w1          = (const float*)d_in[9];
  const float* w2          = (const float*)d_in[10];
  const float* w3          = (const float*)d_in[11];
  const float* cq_norm_w   = (const float*)d_in[12];
  const float* ckv_norm_w  = (const float*)d_in[13];
  const float* cwq         = (const float*)d_in[14];
  const float* cwk         = (const float*)d_in[15];
  const float* cwv         = (const float*)d_in[16];
  const float* cwo         = (const float*)d_in[17];
  const float* patch_proj  = (const float*)d_in[18];

  // ---- workspace layout ----
  char* ws = (char*)d_ws;
  size_t off = 0;
  auto alloc = [&](size_t bytes) -> char* {
    char* p = ws + off;
    off += (bytes + 255) & ~(size_t)255;
    return p;
  };
  float* h        = (float*)alloc((size_t)ROWS_ * D_ * 4);         // residual
  bf16*  xn       = (bf16*) alloc((size_t)ROWS_ * D_ * 2);         // rms out (reused 3x)
  float* qbuf     = (float*)alloc((size_t)ROWS_ * D_ * 4);         // q, later cq
  float* kbuf     = (float*)alloc((size_t)ROWS_ * NKV_ * HD_ * 4); // k, later ck
  float* vbuf     = (float*)alloc((size_t)ROWS_ * NKV_ * HD_ * 4); // v, later cv
  bf16*  aobuf    = (bf16*) alloc((size_t)ROWS_ * D_ * 2);         // attn out, later co
  bf16*  ffnmid   = (bf16*) alloc((size_t)ROWS_ * FFN_ * 2);
  bf16*  pooled   = (bf16*) alloc((size_t)CROWS_ * D_ * 2);        // pooled, later qn
  float* pe       = (float*)alloc((size_t)CROWS_ * D_ * 4);
  bf16*  wqT      = (bf16*) alloc((size_t)D_ * D_ * 2);
  bf16*  wkT      = (bf16*) alloc((size_t)D_ * NKV_ * HD_ * 2);
  bf16*  wvT      = (bf16*) alloc((size_t)D_ * NKV_ * HD_ * 2);
  bf16*  woT      = (bf16*) alloc((size_t)D_ * D_ * 2);
  bf16*  w1T      = (bf16*) alloc((size_t)D_ * FFN_ * 2);
  bf16*  w2T      = (bf16*) alloc((size_t)D_ * FFN_ * 2);
  bf16*  w3T      = (bf16*) alloc((size_t)D_ * FFN_ * 2);
  bf16*  ppT      = (bf16*) alloc((size_t)D_ * D_ * 2);
  bf16*  cwqT     = (bf16*) alloc((size_t)D_ * D_ * 2);
  bf16*  cwkT     = (bf16*) alloc((size_t)D_ * NKV_ * HD_ * 2);
  bf16*  cwvT     = (bf16*) alloc((size_t)D_ * NKV_ * HD_ * 2);
  bf16*  cwoT     = (bf16*) alloc((size_t)D_ * D_ * 2);
  (void)ws_size; (void)in_sizes; (void)n_in; (void)out_size;

  auto T = [&](const float* Wsrc, bf16* Wt, int K, int N) {
    int tot = K * N;
    k_conv_t<<<dim3((tot + 255) / 256), dim3(256), 0, stream>>>(Wsrc, Wt, K, N);
  };
  auto G = [&](int epi, const bf16* A, const bf16* Bt, float* C,
               const float* aux, bf16* obf, int M, int N, int K) {
    dim3 g(N / 64, M / 128);
    switch (epi) {
      case 0: k_gemm<0><<<g, 256, 0, stream>>>(A, Bt, C, aux, obf, M, N, K); break;
      case 1: k_gemm<1><<<g, 256, 0, stream>>>(A, Bt, C, aux, obf, M, N, K); break;
      case 2: k_gemm<2><<<g, 256, 0, stream>>>(A, Bt, C, aux, obf, M, N, K); break;
      default: k_gemm<3><<<g, 256, 0, stream>>>(A, Bt, C, aux, obf, M, N, K); break;
    }
  };

  // ---- weight conversion (fp32 KxN -> bf16 NxK) ----
  T(wq, wqT, D_, D_);          T(wk, wkT, D_, NKV_ * HD_);
  T(wv, wvT, D_, NKV_ * HD_);  T(wo, woT, D_, D_);
  T(w1, w1T, D_, FFN_);        T(w2, w2T, FFN_, D_);
  T(w3, w3T, D_, FFN_);        T(patch_proj, ppT, D_, D_);
  T(cwq, cwqT, D_, D_);        T(cwk, cwkT, D_, NKV_ * HD_);
  T(cwv, cwvT, D_, NKV_ * HD_); T(cwo, cwoT, D_, D_);

  // ---- 1. embedding ----
  k_embed<<<dim3(ROWS_), dim3(256), 0, stream>>>(tokens, tok_emb, h);

  // ---- 2. attn RMSNorm -> xn (bf16) ----
  k_rms<<<dim3(ROWS_), dim3(256), 0, stream>>>(h, attn_norm_w, xn);

  // ---- 3. QKV projections (WMMA) ----
  G(0, xn, wqT, qbuf, nullptr, nullptr, ROWS_, D_, D_);
  G(0, xn, wkT, kbuf, nullptr, nullptr, ROWS_, NKV_ * HD_, D_);
  G(0, xn, wvT, vbuf, nullptr, nullptr, ROWS_, NKV_ * HD_, D_);

  // ---- 4. RoPE on q and k ----
  k_rope<<<dim3(ROWS_ * NH_ / 4), dim3(128), 0, stream>>>(qbuf, NH_, ROWS_ * NH_);
  k_rope<<<dim3(ROWS_ * NKV_ / 4), dim3(128), 0, stream>>>(kbuf, NKV_, ROWS_ * NKV_);

  // ---- 5. sliding-window attention -> aobuf (bf16) ----
  k_swa<<<dim3(NH_, S_ / W_, B_), dim3(128), 0, stream>>>(qbuf, kbuf, vbuf, aobuf);

  // ---- 6. output projection, residual add into h ----
  G(1, aobuf, woT, h, nullptr, nullptr, ROWS_, D_, D_);

  // ---- 7. FFN ----
  k_rms<<<dim3(ROWS_), dim3(256), 0, stream>>>(h, ffn_norm_w, xn);
  k_gemm_ffn<<<dim3(FFN_ / 64, ROWS_ / 128), dim3(256), 0, stream>>>(
      xn, w1T, w3T, ffnmid, ROWS_, FFN_, D_);
  G(1, ffnmid, w2T, h, nullptr, nullptr, ROWS_, D_, FFN_);

  // ---- 8. patch max-pool + projection -> pe ----
  k_pool<<<dim3(CROWS_), dim3(256), 0, stream>>>(h, pooled);
  G(0, pooled, ppT, pe, nullptr, nullptr, CROWS_, D_, D_);

  // ---- 9. cross-attn projections ----
  k_rms<<<dim3(CROWS_), dim3(256), 0, stream>>>(pe, cq_norm_w, pooled);  // qn
  G(0, pooled, cwqT, qbuf, nullptr, nullptr, CROWS_, D_, D_);            // cq
  k_rms<<<dim3(ROWS_), dim3(256), 0, stream>>>(h, ckv_norm_w, xn);       // kvn
  G(0, xn, cwkT, kbuf, nullptr, nullptr, ROWS_, NKV_ * HD_, D_);         // ck
  G(0, xn, cwvT, vbuf, nullptr, nullptr, ROWS_, NKV_ * HD_, D_);         // cv

  // ---- 10. cross attention -> aobuf (co, bf16) ----
  k_xattn<<<dim3(NH_, NP_ / 128, B_), dim3(128), 0, stream>>>(qbuf, kbuf, vbuf, aobuf);

  // ---- 11. final: out = pe + co @ cwo ----
  G(3, aobuf, cwoT, (float*)d_out, pe, nullptr, CROWS_, D_, D_);
}
